// MemoryBank_30099130810384
// MI455X (gfx1250) — matrix operand order repaired and checked by reading, not compiled, not run on hardware
//
#include <hip/hip_runtime.h>

static constexpr int NB  = 32;    // batch
static constexpr int DIM = 128;   // feature dim (K)
static constexpr int T   = 4096;  // time steps
static constexpr int SZ  = 512;   // memory units (N, softmax axis)
static constexpr int APITCH = 136; // bf16 elems per LDS A row (128 + 16B pad)

typedef __attribute__((ext_vector_type(16))) __bf16 v16bf;
typedef __attribute__((ext_vector_type(8)))  __bf16 v8bf;
typedef __attribute__((ext_vector_type(4)))  __bf16 v4bf;
typedef __attribute__((ext_vector_type(8)))  float  v8f;
typedef __attribute__((ext_vector_type(4)))  float  v4f;

// One block = 512 threads = 16 wave32 waves. Wave w owns columns [w*32, w*32+32).
// Block handles 256 consecutive t-rows of one batch (16 M-tiles of 16 rows).
__global__ __launch_bounds__(512)
void memorybank_wmma_kernel(const float* __restrict__ H,
                            const float* __restrict__ U,
                            float* __restrict__ out)
{
    const int tid  = threadIdx.x;
    const int lane = tid & 31;
    const int wv   = tid >> 5;     // 0..15
    const int half = lane >> 4;    // 0: lanes 0-15, 1: lanes 16-31
    const int l16  = lane & 15;

    const int b   = blockIdx.x >> 4;   // batch index
    const int seg = blockIdx.x & 15;   // 256-row segment within batch
    const float* __restrict__ Hb = H + (size_t)b * DIM * T;

    __shared__ __align__(16) __bf16 ldsA[16][APITCH];  // A tile, [t-row][k], bf16
    __shared__ __align__(16) float  redsum[16][16];    // [row-in-tile][wave] partial sums
    __shared__ __align__(16) float  gsum[16];          // per-row softmax denominators

    // ---------- B (units) fragments, resident in VGPRs as bf16 ----------
    // B 32x16 operand, contiguous k order used here (not ISA table order): lanes 0-15 hold K=0..15, lanes 16-31 K=16..31,
    // column N = lane%16, VGPR v packs (K=2v, 2v+1).
    v16bf bfrag[2][4];
    float c2[2];  // (SHIFT - ||units[:,s]||^2) * log2(e), folded exp constant
    const int s_base = wv * 32;
    const float LOG2E = 1.4426950408889634f;
    const float SHIFT = 40.0f;  // softmax is shift-invariant; keeps exp in f32 range
    #pragma unroll
    for (int j = 0; j < 2; ++j) {
        const int s = s_base + j * 16 + l16;
        float a2 = 0.f;
        #pragma unroll
        for (int ks = 0; ks < 4; ++ks) {
            const int kb = ks * 32 + half * 16;
            v16bf f;
            #pragma unroll
            for (int i = 0; i < 16; ++i) {
                float v = U[(kb + i) * SZ + s];
                f[i] = (__bf16)v;
                a2 += v * v;
            }
            bfrag[j][ks] = f;
        }
        // other K-half of the column norm lives in lane^16 (same s)
        float m2 = a2 + __shfl_xor(a2, 16, 32);
        c2[j] = (SHIFT - m2) * LOG2E;
    }

    // ---------- cooperative A staging pattern ----------
    // thread -> (t-col tloc, 4 consecutive k rows); global loads coalesce over tloc.
    const int tloc = tid & 15;
    const int k4   = (tid >> 4) << 2;            // 0,4,...,124
    const float* __restrict__ gp = Hb + (size_t)k4 * T + tloc;

    // prologue: load tile 0 into registers
    int t0 = seg * 256;
    float g0 = gp[0 * T + t0], g1 = gp[1 * T + t0],
          g2 = gp[2 * T + t0], g3 = gp[3 * T + t0];

    for (int it = 0; it < 16; ++it) {
        t0 = seg * 256 + it * 16;

        // ---- stage A tile to LDS as bf16, transposed to [t][k] ----
        v4bf pk = { (__bf16)g0, (__bf16)g1, (__bf16)g2, (__bf16)g3 };
        *(v4bf*)&ldsA[tloc][k4] = pk;            // ds_store_b64
        __syncthreads();

        // issue next tile's global loads early (latency hidden behind WMMA+softmax)
        if (it < 15) {
            const int tn = t0 + 16;
            g0 = gp[0 * T + tn]; g1 = gp[1 * T + tn];
            g2 = gp[2 * T + tn]; g3 = gp[3 * T + tn];
        }
        // speculative prefetch ahead (OOB prefetches are dropped by HW)
        __builtin_prefetch(&gp[(size_t)(it & 3) * T + t0 + 32], 0, 1);

        // ---- preload all A fragments (one ds clause, one wait) ----
        // A 16x32 layout (matches B): lanes 0-15 K 0..15,
        // lanes 16-31 K 16..31; row M = lane%16.
        v8bf alo[4], ahi[4];
        #pragma unroll
        for (int ks = 0; ks < 4; ++ks) {
            const int kb = ks * 32 + half * 16;
            alo[ks] = *(const v8bf*)&ldsA[l16][kb];        // ds_load_b128
            ahi[ks] = *(const v8bf*)&ldsA[l16][kb + 8];    // ds_load_b128
        }

        // ---- GEMM: 8 WMMAs per wave per tile ----
        v8f acc0 = {0.f,0.f,0.f,0.f,0.f,0.f,0.f,0.f};
        v8f acc1 = acc0;
        #pragma unroll
        for (int ks = 0; ks < 4; ++ks) {
            v16bf a = __builtin_shufflevector(alo[ks], ahi[ks],
                        0,1,2,3,4,5,6,7,8,9,10,11,12,13,14,15);
            acc0 = __builtin_amdgcn_wmma_f32_16x16x32_bf16(
                       false, a, false, bfrag[0][ks], (short)0, acc0, false, false);
            acc1 = __builtin_amdgcn_wmma_f32_16x16x32_bf16(
                       false, a, false, bfrag[1][ks], (short)0, acc1, false, false);
        }

        // ---- softmax over s (shift-invariant; constant shift, no max pass) ----
        // exp(2*cross - m2 + SHIFT) = exp2(fma(acc, 2*log2e, c2))
        // C/D layout: VGPR r, lane L -> row M = r + half*8, col N = lane%16.
        float e[2][8];
        float pr[8];
        #pragma unroll
        for (int r = 0; r < 8; ++r) {
            e[0][r] = __builtin_amdgcn_exp2f(fmaf(acc0[r], 2.f * LOG2E, c2[0]));
            e[1][r] = __builtin_amdgcn_exp2f(fmaf(acc1[r], 2.f * LOG2E, c2[1]));
            float s = e[0][r] + e[1][r];
            #pragma unroll
            for (int d = 1; d < 16; d <<= 1)   // xor<16 stays within a 16-lane half
                s += __shfl_xor(s, d, 32);
            pr[r] = s;   // wave-local row sum over its 32 columns
        }
        if (l16 == 0) {
            #pragma unroll
            for (int r = 0; r < 8; ++r) redsum[half * 8 + r][wv] = pr[r];
        }
        __syncthreads();

        // tree-reduce 16 wave partials per row (waves 0-7), broadcast via gsum
        if (tid < 256) {
            const int row = tid >> 4;
            float v = redsum[row][tid & 15];
            v += __shfl_xor(v, 1, 32);
            v += __shfl_xor(v, 2, 32);
            v += __shfl_xor(v, 4, 32);
            v += __shfl_xor(v, 8, 32);
            if ((tid & 15) == 0) gsum[row] = v;
        }
        __syncthreads();

        v4f s_lo = *(const v4f*)&gsum[half * 8];       // broadcast ds_load_b128
        v4f s_hi = *(const v4f*)&gsum[half * 8 + 4];
        float inv[8];
        #pragma unroll
        for (int r = 0; r < 4; ++r) {
            inv[r]     = __builtin_amdgcn_rcpf(s_lo[r]);
            inv[r + 4] = __builtin_amdgcn_rcpf(s_hi[r]);
        }

        // ---- transposed store: out[b][s][t]; lane holds 8 consecutive t per col ----
        #pragma unroll
        for (int j = 0; j < 2; ++j) {
            const int s = s_base + j * 16 + l16;
            float* op = out + ((size_t)b * SZ + s) * T + t0 + half * 8;
            v4f olo = { e[j][0]*inv[0], e[j][1]*inv[1], e[j][2]*inv[2], e[j][3]*inv[3] };
            v4f ohi = { e[j][4]*inv[4], e[j][5]*inv[5], e[j][6]*inv[6], e[j][7]*inv[7] };
            *(v4f*)op       = olo;   // global_store_b128
            *(v4f*)(op + 4) = ohi;
        }
    }
}

extern "C" void kernel_launch(void* const* d_in, const int* in_sizes, int n_in,
                              void* d_out, int out_size, void* d_ws, size_t ws_size,
                              hipStream_t stream)
{
    const float* H = (const float*)d_in[0];   // (B, DIM, T) f32
    const float* U = (const float*)d_in[1];   // (DIM, SIZE) f32
    float* out = (float*)d_out;               // (B, SIZE, T) f32

    dim3 grid(NB * 16);  // 512 workgroups: batch * 16 segments of 256 rows
    dim3 block(512);     // 16 wave32 waves
    hipLaunchKernelGGL(memorybank_wmma_kernel, grid, block, 0, stream, H, U, out);
}
